// TemporalPositionalEncoding_13692355739750
// MI455X (gfx1250) — matrix operand order repaired and checked by reading, not compiled, not run on hardware
//
#include <hip/hip_runtime.h>
#include <stdint.h>

typedef __attribute__((ext_vector_type(16))) _Float16 v16h;
typedef __attribute__((ext_vector_type(8)))  float    v8f;
typedef __attribute__((ext_vector_type(4)))  uint32_t v4u;
typedef __attribute__((ext_vector_type(8)))  int      v8i;
typedef __attribute__((ext_vector_type(4)))  int      v4i;

#define BATCH  4
#define LSEQ   4096
#define DMODEL 256
#define ROWS   16      // query rows per workgroup
#define JT     64      // j (key) chunk per iteration
#define XIS    264     // LDS row stride (halves): 256 + 8 pad (16B -> TDM pad_amount=3)
#define XJS    264
#define SST    72      // S-tile row stride (halves): 64 + 8 pad

#define XH_BYTES ((size_t)BATCH * LSEQ * DMODEL * 2)   // f16 copy of x in d_ws

// ---------------- TDM: 2D f16 tile (rows x 256) -> LDS with padded rows ----
// D# per ISA ch.8: data_size=2B, tensor_dim0=256, tile_dim0=256, tile_dim1=rows,
// dim0 stride=256 elems; pad_enable: +4 DWORDs after every 128 DWORDs (one row)
// so LDS rows land on the 264-half (528B) bank-conflict-free stride.
__device__ __forceinline__ void tdm_load_tile(uint32_t ldsOff, const _Float16* gsrc, int rows) {
  const uint64_t ga = (uint64_t)(uintptr_t)gsrc;
  v4u g0;
  g0[0] = 1u;                                   // count=1 (valid descriptor)
  g0[1] = ldsOff;                               // lds_addr (bytes)
  g0[2] = (uint32_t)ga;                         // global_addr[31:0]
  g0[3] = (uint32_t)((ga >> 32) & 0x01FFFFFFu)  // global_addr[56:32]
          | 0x80000000u;                        // type=2 ("image")
  v8i g1;
  g1[0] = (int)((1u << 16)      // data_size = 2 bytes
              | (1u << 20)      // pad_enable
              | (6u << 22)      // pad_interval: 128 DWORDs (one 512B row)
              | (3u << 25));    // pad_amount: 4 DWORDs (16B)
  g1[1] = (int)(256u << 16);    // tensor_dim0 = 256 (bits 79:48)
  g1[2] = (int)(0xFFFFu << 16); // tensor_dim1 = 65535 (bits 111:80, OOB bound only)
  g1[3] = (int)(256u << 16);    // tile_dim0 = 256 (bits 127:112)
  g1[4] = rows;                 // tile_dim1 (bits 143:128), tile_dim2 = 0
  g1[5] = 256;                  // tensor_dim0_stride = 256 elems (bits 207:160)
  g1[6] = 0;
  g1[7] = 0;                    // tensor_dim1_stride unused (2D tile)
  v4i g2 = {0, 0, 0, 0};
  v4i g3 = {0, 0, 0, 0};
#if __clang_major__ >= 23
  __builtin_amdgcn_tensor_load_to_lds(g0, g1, g2, g3, (v8i)0, 0);
#else
  __builtin_amdgcn_tensor_load_to_lds(g0, g1, g2, g3, 0);
#endif
}

// ---- fragment loaders, contiguous k order (k = 16*hi + s) (wave32) ----------------------------
__device__ __forceinline__ v16h afrag_lds(const _Float16* base, int strideH, int kw, int lane) {
  const int m = lane & 15, hi = lane >> 4;
  const uint32_t* row = (const uint32_t*)(base + m * strideH);
  const int d0 = (kw >> 1) + hi * 8;
  union { v16h v; uint32_t u[8]; } r;
#pragma unroll
  for (int p = 0; p < 4; ++p) { r.u[p] = row[d0 + p]; r.u[4 + p] = row[d0 + 4 + p]; }
  return r.v;
}

__device__ __forceinline__ v16h bfrag_rows(const _Float16* base, int strideH, int rowBase, int kw, int lane) {
  const int nn = lane & 15, hi = lane >> 4;
  const uint32_t* row = (const uint32_t*)(base + (rowBase + nn) * strideH);
  const int d0 = (kw >> 1) + hi * 8;
  union { v16h v; uint32_t u[8]; } r;
#pragma unroll
  for (int p = 0; p < 8; ++p) r.u[p] = row[d0 + p];
  return r.v;
}

__device__ __forceinline__ v16h bfrag_cols(const _Float16* base, int strideH, int rowBase, int col0, int lane) {
  const int nn = lane & 15, hi = lane >> 4;
  union { v16h v; uint32_t u[8]; } r;
#pragma unroll
  for (int p = 0; p < 8; ++p) {
    const int k = rowBase + 2 * p + 16 * hi;
    uint32_t lo = *(const uint16_t*)(base + (size_t)k * strideH + col0 + nn);
    uint32_t hh = *(const uint16_t*)(base + (size_t)(k + 1) * strideH + col0 + nn);
    r.u[p] = lo | (hh << 16);
  }
  return r.v;
}

#define WMMA_F16(A, B, C) \
  __builtin_amdgcn_wmma_f32_16x16x32_f16(false, (A), false, (B), (short)0, (C), false, false)

// ---------------- pre-pass: x f32 -> f16 (once), row squared norms ---------
__global__ __launch_bounds__(256) void cvt_kernel(const float* __restrict__ x,
                                                  _Float16* __restrict__ xh,
                                                  float* __restrict__ sq) {
  const int row = blockIdx.x;                 // 0 .. B*L-1
  const int t   = threadIdx.x;                // 0 .. 255
  const float v = x[(size_t)row * DMODEL + t];
  xh[(size_t)row * DMODEL + t] = (_Float16)v;
  float s = v * v;
#pragma unroll
  for (int off = 1; off < 32; off <<= 1) s += __shfl_xor(s, off, 32);
  __shared__ float ps[8];
  if ((t & 31) == 0) ps[t >> 5] = s;
  __syncthreads();
  if (t == 0) {
    float tot = 0.f;
#pragma unroll
    for (int i = 0; i < 8; ++i) tot += ps[i];
    sq[row] = tot;
  }
}

// ---------------- fused gram / exp / S·X kernel ----------------------------
__global__ __launch_bounds__(128) void tpe_kernel(const float* __restrict__ x,
                                                  const _Float16* __restrict__ xh,
                                                  const float* __restrict__ sq,
                                                  float* __restrict__ out) {
  __shared__ _Float16 Xi[ROWS * XIS];
  __shared__ _Float16 XjBuf[2 * JT * XJS];
  __shared__ _Float16 Sb[ROWS * SST];

  const int tid  = threadIdx.x;
  const int lane = tid & 31;
  const int w    = tid >> 5;
  const int nn   = lane & 15;
  const int hi   = lane >> 4;

  const int b  = blockIdx.x / (LSEQ / ROWS);
  const int i0 = (blockIdx.x % (LSEQ / ROWS)) * ROWS;
  const float*    xb  = x  + (size_t)b * LSEQ * DMODEL;
  const _Float16* xhb = xh + (size_t)b * LSEQ * DMODEL;
  const float*    sqb = sq + (size_t)b * LSEQ;

  // ---- prologue: TDM loads for Xi and Xj chunk 0 (wave 0 issues) ----
  if (tid < 32) {
    tdm_load_tile((uint32_t)(uintptr_t)&Xi[0], xhb + (size_t)i0 * DMODEL, ROWS);
    tdm_load_tile((uint32_t)(uintptr_t)&XjBuf[0], xhb, JT);
    __builtin_amdgcn_s_wait_tensorcnt(0);
  }
  __syncthreads();

  v16h aXi[8];
#pragma unroll
  for (int kk = 0; kk < 8; ++kk) aXi[kk] = afrag_lds(Xi, XIS, kk * 32, lane);

  float sqiR[8];
#pragma unroll
  for (int r = 0; r < 8; ++r) sqiR[r] = sqb[i0 + r + 8 * hi];

  v16h ones;
  { union { v16h v; uint32_t u[8]; } t;
#pragma unroll
    for (int p = 0; p < 8; ++p) t.u[p] = 0x3C003C00u;
    ones = t.v; }

  v8f cAcc[4];
#pragma unroll
  for (int dt = 0; dt < 4; ++dt) cAcc[dt] = 0.0f;
  v8f rsum = 0.0f;

  const float EXPC = -0.72134752044448169f;  // exp(-d/2) = exp2(d * -0.5*log2(e))
  const int NITER = LSEQ / JT;

  for (int k = 0; k < NITER; ++k) {
    const int j0 = k * JT;
    const _Float16* cur = &XjBuf[(k & 1) * (JT * XJS)];

    // issue DMA for next chunk into the other buffer (its readers finished
    // before the barrier that ended iteration k-1)
    if (tid < 32 && k + 1 < NITER)
      tdm_load_tile((uint32_t)(uintptr_t)&XjBuf[((k + 1) & 1) * (JT * XJS)],
                    xhb + (size_t)(j0 + JT) * DMODEL, JT);

    // ---- gram tile: 16 query rows x keys (j0 + w*16 ..), K = 256 ----
    v8f g0 = 0.0f, g1 = 0.0f;
#pragma unroll
    for (int kk = 0; kk < 8; kk += 2) {
      v16h bA = bfrag_rows(cur, XJS, w * 16, kk * 32, lane);
      g0 = WMMA_F16(aXi[kk], bA, g0);
      v16h bB = bfrag_rows(cur, XJS, w * 16, (kk + 1) * 32, lane);
      g1 = WMMA_F16(aXi[kk + 1], bB, g1);
    }
    v8f g = g0 + g1;

    // ---- S = exp(-dist^2/2) -> f16 S tile in LDS ----
    const float sqjN = sqb[j0 + w * 16 + nn];
#pragma unroll
    for (int r = 0; r < 8; ++r) {
      const int m = r + 8 * hi;
      float d = sqiR[r] + sqjN - 2.0f * g[r];
      d = fmaxf(d, 0.0f);
      Sb[m * SST + w * 16 + nn] = (_Float16)exp2f(d * EXPC);
    }
    __syncthreads();  // full 16x64 S visible to all waves

    v16h sa0 = afrag_lds(Sb, SST, 0, lane);
    v16h sa1 = afrag_lds(Sb, SST, 32, lane);

    rsum = WMMA_F16(sa0, ones, rsum);
    rsum = WMMA_F16(sa1, ones, rsum);

#pragma unroll
    for (int dt = 0; dt < 4; ++dt) {
      const int col0 = w * 64 + dt * 16;
      v16h b0 = bfrag_cols(cur, XJS, 0, col0, lane);
      cAcc[dt] = WMMA_F16(sa0, b0, cAcc[dt]);
      v16h b1 = bfrag_cols(cur, XJS, 32, col0, lane);
      cAcc[dt] = WMMA_F16(sa1, b1, cAcc[dt]);
    }

    // next buffer's DMA had the whole iteration to land; drain + release
    if (tid < 32) __builtin_amdgcn_s_wait_tensorcnt(0);
    __syncthreads();
  }

  // ---- epilogue: out = x + pe + sem/rowsum (native sin/cos) ----
  const float PEF = -0.051905126482615037f;  // -log2(10000)/256
#pragma unroll
  for (int dt = 0; dt < 4; ++dt) {
#pragma unroll
    for (int r = 0; r < 8; ++r) {
      const int m    = r + 8 * hi;
      const int grow = i0 + m;
      const int col  = w * 64 + dt * 16 + nn;
      float sem = cAcc[dt][r] / rsum[r];
      float xv  = xb[(size_t)grow * DMODEL + col];
      const int c2 = col & ~1;
      float freq = exp2f((float)c2 * PEF);
      float ang  = (float)grow * freq;
      float pe   = (col & 1) ? __cosf(ang) : __sinf(ang);
      out[((size_t)b * LSEQ + grow) * DMODEL + col] = xv + pe + sem;
    }
  }
}

extern "C" void kernel_launch(void* const* d_in, const int* in_sizes, int n_in,
                              void* d_out, int out_size, void* d_ws, size_t ws_size,
                              hipStream_t stream) {
  const float* x = (const float*)d_in[0];
  float* out = (float*)d_out;
  _Float16* xh = (_Float16*)d_ws;
  float* sqws = (float*)((char*)d_ws + XH_BYTES);

  hipLaunchKernelGGL(cvt_kernel, dim3(BATCH * LSEQ), dim3(DMODEL), 0, stream, x, xh, sqws);
  hipLaunchKernelGGL(tpe_kernel, dim3(BATCH * (LSEQ / ROWS)), dim3(128), 0, stream,
                     x, xh, sqws, out);
}
